// GraphRUNet_54640573939801
// MI455X (gfx1250) — matrix operand order repaired and checked by reading, not compiled, not run on hardware
//
#include <hip/hip_runtime.h>

// ---------------- problem constants ----------------
#define NN 50000
#define EE 800000
#define CC 128
#define K1 25000   // ceil(0.5*N)
#define K2 12500
#define K3 6250

// ---------------- WMMA types (CDNA5 / gfx1250, wave32) ----------------
typedef __attribute__((ext_vector_type(16))) __bf16 bf16x16;
typedef __attribute__((ext_vector_type(8)))  float  f32x8;

// TDM descriptor group types (6-arg builtin form: g0,g1,g2,g3,extra,cpol)
typedef __attribute__((ext_vector_type(4))) unsigned u32x4;
typedef __attribute__((ext_vector_type(8))) int      i32x8;
typedef __attribute__((ext_vector_type(4))) int      i32x4;

#define USE_TDM __has_builtin(__builtin_amdgcn_tensor_load_to_lds)

union Frag16 {
    bf16x16  v;
    unsigned u[8];
    uint4    q[2];
};

__device__ __forceinline__ unsigned pack_bf16(float lo, float hi) {
    unsigned a = __float_as_uint(lo);
    unsigned b = __float_as_uint(hi);
    a = (a + 0x7FFFu + ((a >> 16) & 1u)) >> 16;   // RNE truncate to bf16
    b = (b + 0x7FFFu + ((b >> 16) & 1u)) >> 16;
    return (b << 16) | (a & 0xFFFFu);
}

// ---------------- weight prep ----------------------------------------------
// fp32 W[k][n] -> bf16-pair-packed W^T [n][k/2], XOR-swizzled IN GLOBAL MEMORY
// so the TDM (or a plain linear copy) lands a bank-conflict-free LDS image.
// 7 matrices: m=0 -> W_down0, m=1..3 -> W_down[i], m=4..6 -> W_up[i]
__global__ void wprep_kernel(const float* __restrict__ Wd0,
                             const float* __restrict__ Wd,
                             const float* __restrict__ Wu,
                             unsigned* __restrict__ wt) {
    int id = blockIdx.x * blockDim.x + threadIdx.x;
    if (id >= 7 * CC * (CC / 2)) return;
    int m  = id / (CC * (CC / 2));
    int r  = id % (CC * (CC / 2));
    int nn = r >> 6;        // output channel (column of W)
    int kp = r & 63;        // k pair index
    const float* W = (m == 0) ? Wd0
                   : (m <= 3) ? (Wd + (size_t)(m - 1) * CC * CC)
                              : (Wu + (size_t)(m - 4) * CC * CC);
    float lo = W[(size_t)(2 * kp)     * CC + nn];
    float hi = W[(size_t)(2 * kp + 1) * CC + nn];
    // pre-swizzle: groups of 4 uints XORed by column -> conflict-free ds_load_b128
    int dst = m * (CC * 64) + nn * 64 + (kp ^ ((nn & 15) << 2));
    wt[dst] = pack_bf16(lo, hi);
}

// ---------------- GEMM: H[n,128] = X[n,128] @ W[128,128] -------------------
// v_wmma_f32_16x16x32_bf16; one block = 8 waves, each wave owns a 16-row strip.
// W^T tile (32 KB) staged into LDS by the Tensor Data Mover (TENSORcnt).
__global__ __launch_bounds__(256) void gemm128_wmma_bf16(
    const float* __restrict__ X, const unsigned* __restrict__ Wt,
    float* __restrict__ H, int n)
{
    __shared__ unsigned wlds[CC * 64];   // 32 KB of the 320 KB WGP LDS

#if USE_TDM
    if (threadIdx.x < 32) {
        // Wave 0 issues one TDM DMA: 32 KB contiguous global -> LDS.
        // D# group0: count=1 | lds_addr | global_addr[56:0] | type=2
        typedef __attribute__((address_space(3))) unsigned lds_u32;
        unsigned ldsoff = (unsigned)(size_t)((lds_u32*)wlds);
        unsigned long long ga = (unsigned long long)(size_t)Wt;
        u32x4 g0;
        g0.x = 1u;                                             // valid, 1 descriptor
        g0.y = ldsoff;                                         // lds_addr (bytes)
        g0.z = (unsigned)(ga & 0xFFFFFFFFu);                   // global_addr[31:0]
        g0.w = (unsigned)((ga >> 32) & 0x1FFFFFFu) | (2u << 30); // [56:32] | type=2
        // D# group1: data_size=8B; tensor_dim0=tile_dim0=4096 (8B units) = 32 KB;
        // tensor_dim1=tile_dim1=1; dim0_stride=4096; no mask/pad/iterate.
        i32x8 g1;
        g1[0] = 3 << 16;                 // data_size=3 (8B)
        g1[1] = (4096 & 0xFFFF) << 16;   // tensor_dim0[15:0] @ bits 63:48
        g1[2] = 1 << 16;                 // tensor_dim1=1 @ bits 111:80
        g1[3] = (4096 & 0xFFFF) << 16;   // tile_dim0 @ bits 127:112
        g1[4] = 1;                       // tile_dim1=1, tile_dim2=0
        g1[5] = 4096;                    // tensor_dim0_stride[31:0]
        g1[6] = 0;
        g1[7] = 0;
        i32x4 z4 = {0, 0, 0, 0};                 // groups 2/3 unused (<=2D tensor)
        i32x8 z8 = {0, 0, 0, 0, 0, 0, 0, 0};     // extra group (clang-23 form)
        __builtin_amdgcn_tensor_load_to_lds(g0, g1, z4, z4, z8, 0);
        __builtin_amdgcn_s_wait_tensorcnt((short)0);           // s_wait_tensorcnt 0
    }
    __syncthreads();
#else
    for (int i = threadIdx.x; i < CC * 64; i += 256) wlds[i] = Wt[i];
    __syncthreads();
#endif

    const int wave = threadIdx.x >> 5;
    const int lane = threadIdx.x & 31;
    const int half = lane >> 4;
    const int l16  = lane & 15;

    const int  arow = blockIdx.x * 128 + wave * 16 + l16;   // A-matrix row this lane feeds
    const bool inb  = arow < n;
    const float* xr = X + (size_t)arow * CC;
    if (inb) __builtin_prefetch(xr, 0, 3);                  // global_prefetch_b8

    f32x8 acc[8] = {};   // 8 N-tiles of 16 -> full 128-wide output strip

    #pragma unroll
    for (int kk = 0; kk < 4; ++kk) {                        // K = 128 in steps of 32
        // ---- A fragment: 16-bit A 16x32 layout (ISA 7.12.2):
        // lanes 0-15 hold K=[0..7]+[16..23], lanes 16-31 hold +8.
        Frag16 a;
        float4 f0 = {0.f, 0.f, 0.f, 0.f}, f1 = f0, f2 = f0, f3 = f0;
        const int kb = kk * 32 + half * 8;
        if (inb) {
            f0 = *(const float4*)(xr + kb);
            f1 = *(const float4*)(xr + kb + 4);
            f2 = *(const float4*)(xr + kb + 16);
            f3 = *(const float4*)(xr + kb + 20);
        }
        a.u[0] = pack_bf16(f0.x, f0.y); a.u[1] = pack_bf16(f0.z, f0.w);
        a.u[2] = pack_bf16(f1.x, f1.y); a.u[3] = pack_bf16(f1.z, f1.w);
        a.u[4] = pack_bf16(f2.x, f2.y); a.u[5] = pack_bf16(f2.z, f2.w);
        a.u[6] = pack_bf16(f3.x, f3.y); a.u[7] = pack_bf16(f3.z, f3.w);

        #pragma unroll
        for (int t = 0; t < 8; ++t) {
            // ---- B fragment: lanes 0-15 hold K=[0..7]+[16..23] of the 32-block, lanes 16-31 hold +8.
            Frag16 b;
            const int nb = t * 16 + l16;                    // output column
            const int sw = (nb & 15) << 2;
            const unsigned* base = wlds + nb * 64;
            const int o0 = kk * 16 + half * 4;
            b.q[0] = *(const uint4*)(base + ( o0      ^ sw));
            b.q[1] = *(const uint4*)(base + ((o0 + 8) ^ sw));
            acc[t] = __builtin_amdgcn_wmma_f32_16x16x32_bf16(
                false, a.v, false, b.v, (short)0, acc[t], false, false);
        }
    }

    // ---- C/D layout: VGPR v, lanes 0-15 -> M=v, lanes 16-31 -> M=v+8, N=lane&15.
    const int rbase = blockIdx.x * 128 + wave * 16 + half * 8;
    #pragma unroll
    for (int t = 0; t < 8; ++t) {
        #pragma unroll
        for (int v = 0; v < 8; ++v) {
            int rr = rbase + v;
            if (rr < n) H[(size_t)rr * CC + t * 16 + l16] = acc[t][v];
        }
    }
}

// ---------------- elementwise / graph kernels ----------------
__global__ void fill_f32_kernel(float* p, float v, int n) {
    int i = blockIdx.x * blockDim.x + threadIdx.x;
    if (i < n) p[i] = v;
}

__global__ void copy_f32_kernel(const float* __restrict__ a, float* __restrict__ b, int n) {
    int i = blockIdx.x * blockDim.x + threadIdx.x;
    if (i < n) b[i] = a[i];
}

__global__ void relu_kernel(float* p, int n) {
    int i = blockIdx.x * blockDim.x + threadIdx.x;
    if (i < n) p[i] = fmaxf(p[i], 0.f);
}

__global__ void deg_accum_kernel(const int* __restrict__ src, const int* __restrict__ dst,
                                 const float* __restrict__ w, float* __restrict__ deg, int e) {
    int i = blockIdx.x * blockDim.x + threadIdx.x;
    if (i < e) {
        float we = w[i];
        if (we != 0.f) atomicAdd(&deg[dst[i]], we);
    }
    (void)src;
}

__global__ void rsqrt_kernel(const float* __restrict__ deg, float* __restrict__ dinv, int n) {
    int i = blockIdx.x * blockDim.x + threadIdx.x;
    if (i < n) dinv[i] = rsqrtf(deg[i]);
}

// out = h * dinv^2 (per row) + bias   (self-loop term + bias)
__global__ void self_bias_kernel(const float* __restrict__ h, const float* __restrict__ dinv,
                                 const float* __restrict__ b, float* __restrict__ out, int n) {
    int i = blockIdx.x * blockDim.x + threadIdx.x;
    if (i < n * CC) {
        int r = i >> 7, c = i & 127;
        float di = dinv[r];
        out[i] = h[i] * di * di + b[c];
    }
}

// one block per edge, one thread per channel: out[dst] += h[src] * dinv[s]*w*dinv[d]
__global__ void edge_scatter_kernel(const int* __restrict__ src, const int* __restrict__ dst,
                                    const float* __restrict__ w, const float* __restrict__ dinv,
                                    const float* __restrict__ h, float* __restrict__ out) {
    int e  = blockIdx.x;
    float we = w[e];
    if (we == 0.f) return;
    int s = src[e], d = dst[e];
    float nm = dinv[s] * we * dinv[d];
    int c = threadIdx.x;
    atomicAdd(&out[(size_t)d * CC + c], h[(size_t)s * CC + c] * nm);
}

// score = tanh((x . p) / ||p||)
__global__ void score_kernel(const float* __restrict__ x, const float* __restrict__ p,
                             float* __restrict__ sc, int n) {
    int i = blockIdx.x * blockDim.x + threadIdx.x;
    if (i >= n) return;
    const float* xr = x + (size_t)i * CC;
    float s = 0.f, q = 0.f;
    for (int c = 0; c < CC; ++c) {
        float pv = p[c];
        s += xr[c] * pv;
        q += pv * pv;
    }
    sc[i] = tanhf(s * rsqrtf(q));
}

// stable descending rank (matches lax.top_k tie-breaking): LDS-tiled O(n^2)
__global__ __launch_bounds__(256) void rank_kernel(const float* __restrict__ score,
                                                   int* __restrict__ rank, int n) {
    __shared__ float sh[256];
    int i = blockIdx.x * 256 + threadIdx.x;
    float si = (i < n) ? score[i] : 0.f;
    int cnt = 0;
    for (int base = 0; base < n; base += 256) {
        int j = base + threadIdx.x;
        sh[threadIdx.x] = (j < n) ? score[j] : 0.f;
        __syncthreads();
        int lim = min(256, n - base);
        for (int t = 0; t < lim; ++t) {
            float sj = sh[t];
            int jj = base + t;
            cnt += (sj > si) || (sj == si && jj < i);
        }
        __syncthreads();
    }
    if (i < n) rank[i] = cnt;
}

__global__ void build_perm_kernel(const int* __restrict__ rank, int* __restrict__ perm,
                                  int* __restrict__ remap, int n, int k) {
    int i = blockIdx.x * blockDim.x + threadIdx.x;
    if (i >= n) return;
    int r = rank[i];
    if (r < k) { perm[r] = i; remap[i] = r; }
    else       { remap[i] = -1; }
}

// x_new[r] = x[perm[r]] * score[perm[r]]
__global__ void gather_scale_kernel(const float* __restrict__ xi, const int* __restrict__ perm,
                                    const float* __restrict__ sc, float* __restrict__ xo) {
    int r = blockIdx.x, c = threadIdx.x;
    int i = perm[r];
    xo[(size_t)r * CC + c] = xi[(size_t)i * CC + c] * sc[i];
}

__global__ void remap_edges_kernel(const int* __restrict__ s, const int* __restrict__ d,
                                   const float* __restrict__ w, const int* __restrict__ remap,
                                   int* __restrict__ so, int* __restrict__ dd,
                                   float* __restrict__ wo, int e) {
    int i = blockIdx.x * blockDim.x + threadIdx.x;
    if (i >= e) return;
    int ns = remap[s[i]], nd = remap[d[i]];
    float we = w[i];
    bool valid = (ns >= 0) && (nd >= 0) && (we > 0.f);
    so[i] = valid ? ns : 0;
    dd[i] = valid ? nd : 0;
    wo[i] = valid ? we : 0.f;
}

// t[perm[r]] += xb[r]   (perm injective -> no atomics needed)
__global__ void unpool_add_kernel(float* __restrict__ t, const int* __restrict__ perm,
                                  const float* __restrict__ xb) {
    int r = blockIdx.x, c = threadIdx.x;
    t[(size_t)perm[r] * CC + c] += xb[(size_t)r * CC + c];
}

// ---------------- host orchestration ----------------
extern "C" void kernel_launch(void* const* d_in, const int* in_sizes, int n_in,
                              void* d_out, int out_size, void* d_ws, size_t ws_size,
                              hipStream_t stream) {
    (void)in_sizes; (void)n_in; (void)out_size; (void)ws_size;

    const float* x_in = (const float*)d_in[0];
    const int*   eidx = (const int*)d_in[1];
    // d_in[2] = edge_type (unused by the reference)
    const float* Wd0  = (const float*)d_in[3];
    const float* bd0  = (const float*)d_in[4];
    const float* Wd   = (const float*)d_in[5];
    const float* bd   = (const float*)d_in[6];
    const float* pp   = (const float*)d_in[7];
    const float* Wu   = (const float*)d_in[8];
    const float* bu   = (const float*)d_in[9];

    const int* src0 = eidx;
    const int* dst0 = eidx + EE;

    // ---- bump-allocate scratch from d_ws ----
    char*  ws  = (char*)d_ws;
    size_t off = 0;
    auto alloc = [&](size_t bytes) -> char* {
        char* p = ws + off;
        off += (bytes + 255) & ~(size_t)255;
        return p;
    };

    unsigned* wt_all = (unsigned*)alloc(7 * CC * 64 * sizeof(unsigned));
    float* w_ones = (float*)alloc((size_t)EE * 4);
    int*   src1 = (int*)alloc((size_t)EE * 4);
    int*   dst1 = (int*)alloc((size_t)EE * 4);
    float* w1   = (float*)alloc((size_t)EE * 4);
    int*   src2 = (int*)alloc((size_t)EE * 4);
    int*   dst2 = (int*)alloc((size_t)EE * 4);
    float* w2   = (float*)alloc((size_t)EE * 4);
    int*   src3 = (int*)alloc((size_t)EE * 4);
    int*   dst3 = (int*)alloc((size_t)EE * 4);
    float* w3   = (float*)alloc((size_t)EE * 4);
    float* xA   = (float*)alloc((size_t)NN * CC * 4);   // xs[0]
    float* xB   = (float*)alloc((size_t)K1 * CC * 4);   // xs[1]
    float* xC   = (float*)alloc((size_t)K2 * CC * 4);   // xs[2]
    float* xD   = (float*)alloc((size_t)K3 * CC * 4);   // bottom
    float* xP   = (float*)alloc((size_t)NN * CC * 4);   // generic conv-input scratch
    float* hBuf = (float*)alloc((size_t)NN * CC * 4);   // GEMM output
    float* deg  = (float*)alloc((size_t)NN * 4);
    float* dinv = (float*)alloc((size_t)NN * 4);
    float* scbf = (float*)alloc((size_t)NN * 4);
    int*   rank = (int*)alloc((size_t)NN * 4);
    int*   rmap = (int*)alloc((size_t)NN * 4);
    int*   prm1 = (int*)alloc((size_t)K1 * 4);
    int*   prm2 = (int*)alloc((size_t)K2 * 4);
    int*   prm3 = (int*)alloc((size_t)K3 * 4);

    // ---- one-time per launch: pack weights to bf16 W^T (pre-swizzled), w = 1 ----
    wprep_kernel<<<(7 * CC * 64 + 255) / 256, 256, 0, stream>>>(Wd0, Wd, Wu, wt_all);
    fill_f32_kernel<<<(EE + 255) / 256, 256, 0, stream>>>(w_ones, 1.0f, EE);

    auto run_gcn = [&](const float* xin, const int* src, const int* dst, const float* w,
                       int n, int wmat, const float* bias, float* out, bool relu_on) {
        fill_f32_kernel<<<(n + 255) / 256, 256, 0, stream>>>(deg, 1.0f, n);
        deg_accum_kernel<<<(EE + 255) / 256, 256, 0, stream>>>(src, dst, w, deg, EE);
        rsqrt_kernel<<<(n + 255) / 256, 256, 0, stream>>>(deg, dinv, n);
        gemm128_wmma_bf16<<<(n + 127) / 128, 256, 0, stream>>>(xin, wt_all + (size_t)wmat * CC * 64, hBuf, n);
        self_bias_kernel<<<(n * CC + 255) / 256, 256, 0, stream>>>(hBuf, dinv, bias, out, n);
        edge_scatter_kernel<<<EE, CC, 0, stream>>>(src, dst, w, dinv, hBuf, out);
        if (relu_on) relu_kernel<<<(n * CC + 255) / 256, 256, 0, stream>>>(out, n * CC);
    };

    auto run_pool = [&](const float* xin, int n, int k, const float* p,
                        const int* sI, const int* dI, const float* wI,
                        int* sO, int* dO, float* wO, int* perm, float* xO) {
        score_kernel<<<(n + 255) / 256, 256, 0, stream>>>(xin, p, scbf, n);
        rank_kernel<<<(n + 255) / 256, 256, 0, stream>>>(scbf, rank, n);
        build_perm_kernel<<<(n + 255) / 256, 256, 0, stream>>>(rank, perm, rmap, n, k);
        gather_scale_kernel<<<k, CC, 0, stream>>>(xin, perm, scbf, xO);
        remap_edges_kernel<<<(EE + 255) / 256, 256, 0, stream>>>(sI, dI, wI, rmap, sO, dO, wO, EE);
    };

    // ---------------- down path ----------------
    run_gcn(x_in, src0, dst0, w_ones, NN, 0, bd0, xA, true);                 // xs[0]

    run_pool(xA, NN, K1, pp + 0 * CC, src0, dst0, w_ones, src1, dst1, w1, prm1, xP);
    run_gcn(xP, src1, dst1, w1, K1, 1, bd + 0 * CC, xB, true);              // xs[1]

    run_pool(xB, K1, K2, pp + 1 * CC, src1, dst1, w1, src2, dst2, w2, prm2, xP);
    run_gcn(xP, src2, dst2, w2, K2, 2, bd + 1 * CC, xC, true);              // xs[2]

    run_pool(xC, K2, K3, pp + 2 * CC, src2, dst2, w2, src3, dst3, w3, prm3, xP);
    run_gcn(xP, src3, dst3, w3, K3, 3, bd + 2 * CC, xD, true);              // bottom

    // ---------------- up path ----------------
    // i=0 (level 2): x = xs[2] + unpool(xD, perm3); conv W_up[0]; relu
    copy_f32_kernel<<<(K2 * CC + 255) / 256, 256, 0, stream>>>(xC, xP, K2 * CC);
    unpool_add_kernel<<<K3, CC, 0, stream>>>(xP, prm3, xD);
    run_gcn(xP, src2, dst2, w2, K2, 4, bu + 0 * CC, xC, true);

    // i=1 (level 1): x = xs[1] + unpool(xC, perm2); conv W_up[1]; relu
    copy_f32_kernel<<<(K1 * CC + 255) / 256, 256, 0, stream>>>(xB, xP, K1 * CC);
    unpool_add_kernel<<<K2, CC, 0, stream>>>(xP, prm2, xC);
    run_gcn(xP, src1, dst1, w1, K1, 5, bu + 1 * CC, xB, true);

    // i=2 (level 0): x = xs[0] + unpool(xB, perm1); conv W_up[2]; NO relu -> d_out
    copy_f32_kernel<<<(NN * CC + 255) / 256, 256, 0, stream>>>(xA, xP, NN * CC);
    unpool_add_kernel<<<K1, CC, 0, stream>>>(xP, prm1, xB);
    run_gcn(xP, src0, dst0, w_ones, NN, 6, bu + 2 * CC, (float*)d_out, false);
}
